// LocalAttention_42399917146588
// MI455X (gfx1250) — matrix operand order repaired and checked by reading, not compiled, not run on hardware
//
#include <hip/hip_runtime.h>

// ---------------------------------------------------------------------------
// Local windowed attention on MI455X (gfx1250), bf16 WMMA + f32 accumulate.
// Pipeline: cvt(x), transpose-cvt(w_qkv, w_out) -> GEMM(qkv) -> flash-style
// windowed attention -> GEMM(out)+bias.
// GEMM uses TDM (tensor_load_to_lds) double-buffered tile staging when the
// toolchain exposes the builtin; otherwise falls back to direct global
// fragment loads (round-1 path, known to compile).
// ---------------------------------------------------------------------------

static constexpr int   kNTok  = 16384;
static constexpr int   kDim   = 1024;
static constexpr int   kHeads = 8;
static constexpr int   kDHead = 64;
static constexpr int   kWsz   = 128;
static constexpr int   kNWin  = kNTok / kWsz;    // 128
static constexpr int   kHD    = kHeads * kDHead; // 512
static constexpr int   kQKV   = 3 * kHD;         // 1536
static constexpr float kScale = 0.125f;          // 64^-0.5
#define ATTN_MASK_VALUE (-1e10f)

typedef __attribute__((ext_vector_type(16))) __bf16   bf16x16;
typedef __attribute__((ext_vector_type(8)))  float    f32x8;
typedef __attribute__((ext_vector_type(4)))  unsigned u32x4;
typedef __attribute__((ext_vector_type(4)))  unsigned uint32x4;
typedef __attribute__((ext_vector_type(8)))  int      int32x8;
typedef __attribute__((ext_vector_type(4)))  int      int32x4;

union BFragU { bf16x16 v; u32x4 q[2]; };

// A fragment (16x32 bf16, M in lanes): lanes<16 hold K 0..15,
// lanes>=16 hold K 16..31 (same K order as load_b_frag).
__device__ __forceinline__ bf16x16 load_a_frag(const __bf16* row, int k, int hi) {
  BFragU f;
  f.q[0] = *(const u32x4*)(row + k + hi * 16);
  f.q[1] = *(const u32x4*)(row + k + hi * 16 + 8);
  return f.v;
}

// B fragment (32x16 bf16, N in lanes): lanes<16 hold K 0..15, lanes>=16 hold
// K 16..31; colp points at the (transposed) K-contiguous column.
__device__ __forceinline__ bf16x16 load_b_frag(const __bf16* colp, int k, int hi) {
  BFragU f;
  f.q[0] = *(const u32x4*)(colp + k + hi * 16);
  f.q[1] = *(const u32x4*)(colp + k + hi * 16 + 8);
  return f.v;
}

__device__ __forceinline__ f32x8 wmma_bf16(bf16x16 a, bf16x16 b, f32x8 c) {
  return __builtin_amdgcn_wmma_f32_16x16x32_bf16(false, a, false, b, (short)0, c,
                                                 false, false);
}

// ---------------------------------------------------------------------------
// TDM support (device pass only; host pass takes the fallback GEMM body).
// ---------------------------------------------------------------------------
#if __has_builtin(__builtin_amdgcn_tensor_load_to_lds)
#define HAS_TDM 1
#if __has_include(<hip/amd_detail/amd_gfx1250_TDM.h>)
// amdgpu-toolchain (clang-23 / therock): 6-arg builtin
#define TDM_LOAD(g0, g1) \
  __builtin_amdgcn_tensor_load_to_lds((g0), (g1), (int32x4)0, (int32x4)0, (int32x8)0, 0)
#else
// ROCm 7.2 (clang-22): 5-arg builtin
#define TDM_LOAD(g0, g1) \
  __builtin_amdgcn_tensor_load_to_lds((g0), (g1), (int32x4)0, (int32x4)0, 0)
#endif

#if __has_builtin(__builtin_amdgcn_s_wait_tensorcnt)
#define WAIT_TENSOR() __builtin_amdgcn_s_wait_tensorcnt(0)
#else
#define WAIT_TENSOR() asm volatile("s_wait_tensorcnt 0x0" ::: "memory")
#endif

// Generic pointers to LDS carry the wave-relative LDS byte offset in their low
// 32 bits.
__device__ __forceinline__ unsigned lds_byte_off(const void* p) {
  return (unsigned)(__SIZE_TYPE__)p;
}

// Issue one TDM 2D tile load: tile = 32 (K, contiguous) x 128 (rows), bf16,
// row pitch = strideElems. D# packing per ISA 8.3/8.4.
__device__ __forceinline__ void tdm_load_tile_32x128(const __bf16* gsrc,
                                                     unsigned ldsOff,
                                                     int strideElems) {
  unsigned long long ga = (unsigned long long)(__SIZE_TYPE__)gsrc;
  uint32x4 g0;
  g0[0] = 1u;                                          // count=1, no gather
  g0[1] = ldsOff;                                      // lds_addr (bytes)
  g0[2] = (unsigned)(ga & 0xffffffffull);              // global_addr[31:0]
  g0[3] = (unsigned)((ga >> 32) & 0x1ffffffull)        // global_addr[56:32]
          | 0x80000000u;                               // type=2 ("image")
  int32x8 g1;
  g1[0] = 0x10000;                 // workgroup_mask=0, data_size=1 (2 bytes)
  g1[1] = (int)(32u << 16);        // tensor_dim0[15:0]=32 (in [63:48])
  g1[2] = (int)(128u << 16);       // tensor_dim0 hi=0 | tensor_dim1[15:0]=128
  g1[3] = (int)(32u << 16);        // tensor_dim1 hi=0 | tile_dim0=32
  g1[4] = 128;                     // tile_dim1=128 | tile_dim2=0
  g1[5] = strideElems;             // tensor_dim0_stride[31:0]
  g1[6] = 0;                       // stride0 hi | stride1 lo
  g1[7] = 0;
  TDM_LOAD(g0, g1);
}
#endif  // HAS_TDM

// ---------------------------------------------------------------------------
__global__ void cvt_f32_to_bf16(const float* __restrict__ in, __bf16* __restrict__ out,
                                int n) {
  for (int i = blockIdx.x * blockDim.x + threadIdx.x; i < n; i += gridDim.x * blockDim.x)
    out[i] = (__bf16)in[i];
}

// in: K x N row-major f32  ->  out: N x K row-major bf16 (transpose + convert)
__global__ void transpose_f32_to_bf16(const float* __restrict__ in,
                                      __bf16* __restrict__ out, int K, int N) {
  int total = K * N;
  for (int i = blockIdx.x * blockDim.x + threadIdx.x; i < total;
       i += gridDim.x * blockDim.x) {
    int k = i / N;
    int n2 = i - k * N;
    out[(size_t)n2 * K + k] = (__bf16)in[i];
  }
}

// ---------------------------------------------------------------------------
// GEMM: C[M,N] = A[M,K] * BT[N,K]^T, bf16 inputs, f32 accumulate.
// Block = 256 threads = 8 waves (4 M x 2 N); wave tile 32x64 (2x4 subtiles).
// ---------------------------------------------------------------------------
#if defined(HAS_TDM)

template <bool F32OUT>
__global__ __launch_bounds__(256) void gemm_bf16_wmma(
    const __bf16* __restrict__ A, const __bf16* __restrict__ BT,
    const float* __restrict__ bias, void* __restrict__ outp, int N, int K) {
  const int tid = threadIdx.x;
  const int lane = tid & 31;
  const int wv = tid >> 5;   // 0..7
  const int wm = wv >> 1;    // 0..3
  const int wn = wv & 1;     // 0..1
  const int hi = lane >> 4;
  const int col = lane & 15;
  const int mBase0 = blockIdx.y * 128;
  const int nBase0 = blockIdx.x * 128;

  // Double-buffered tiles: A rows [mBase0,+128) x K-chunk 32, row-major 128x32;
  // B rows (=N cols) [nBase0,+128) x K-chunk 32. 8KB each, 32KB total.
  __shared__ __align__(16) __bf16 Asm[2][128 * 32];
  __shared__ __align__(16) __bf16 Bsm[2][128 * 32];

  f32x8 acc[2][4];
#pragma unroll
  for (int mi = 0; mi < 2; ++mi)
#pragma unroll
    for (int ni = 0; ni < 4; ++ni)
#pragma unroll
      for (int e = 0; e < 8; ++e) acc[mi][ni][e] = 0.0f;

  // Wave 0 drives the TDM; TENSORcnt is per-wave, so only wave 0 waits and
  // the block barrier publishes the LDS tiles to the other waves.
  if (wv == 0) {
    tdm_load_tile_32x128(A + (size_t)mBase0 * K, lds_byte_off(&Asm[0][0]), K);
    tdm_load_tile_32x128(BT + (size_t)nBase0 * K, lds_byte_off(&Bsm[0][0]), K);
    WAIT_TENSOR();
  }
  __syncthreads();

  const int T = K >> 5;  // K/32 chunks
  int cur = 0;
  for (int t = 0; t < T; ++t) {
    // Kick off the next chunk's DMA before consuming the current one.
    if (t + 1 < T && wv == 0) {
      const int kn = (t + 1) << 5;
      tdm_load_tile_32x128(A + (size_t)mBase0 * K + kn,
                           lds_byte_off(&Asm[cur ^ 1][0]), K);
      tdm_load_tile_32x128(BT + (size_t)nBase0 * K + kn,
                           lds_byte_off(&Bsm[cur ^ 1][0]), K);
    }

    bf16x16 af[2], bfr[4];
#pragma unroll
    for (int mi = 0; mi < 2; ++mi) {
      const __bf16* aRow = &Asm[cur][(wm * 32 + mi * 16 + col) * 32];
      af[mi] = load_a_frag(aRow, 0, hi);
    }
#pragma unroll
    for (int ni = 0; ni < 4; ++ni) {
      const __bf16* bRow = &Bsm[cur][(wn * 64 + ni * 16 + col) * 32];
      bfr[ni] = load_b_frag(bRow, 0, hi);
    }
#pragma unroll
    for (int mi = 0; mi < 2; ++mi)
#pragma unroll
      for (int ni = 0; ni < 4; ++ni)
        acc[mi][ni] = wmma_bf16(af[mi], bfr[ni], acc[mi][ni]);

    if (t + 1 < T) {
      if (wv == 0) WAIT_TENSOR();   // next buffer landed
      __syncthreads();              // ... and everyone is done reading cur
      cur ^= 1;
    }
  }

  // C/D layout: VGPR i -> row (i + 8*hi), col = lane&15 (ISA 7.12.2).
#pragma unroll
  for (int mi = 0; mi < 2; ++mi)
#pragma unroll
    for (int ni = 0; ni < 4; ++ni)
#pragma unroll
      for (int i = 0; i < 8; ++i) {
        int row = mBase0 + wm * 32 + mi * 16 + i + 8 * hi;
        int c2 = nBase0 + wn * 64 + ni * 16 + col;
        float v = acc[mi][ni][i];
        if (F32OUT)
          ((float*)outp)[(size_t)row * N + c2] = v + bias[c2];
        else
          ((__bf16*)outp)[(size_t)row * N + c2] = (__bf16)v;
      }
}

#else  // !HAS_TDM: direct-global fragment loads (round-1 path)

template <bool F32OUT>
__global__ __launch_bounds__(256) void gemm_bf16_wmma(
    const __bf16* __restrict__ A, const __bf16* __restrict__ BT,
    const float* __restrict__ bias, void* __restrict__ outp, int N, int K) {
  const int tid = threadIdx.x;
  const int lane = tid & 31;
  const int wv = tid >> 5;
  const int wm = wv >> 1;
  const int wn = wv & 1;
  const int hi = lane >> 4;
  const int col = lane & 15;
  const int mBase = blockIdx.y * 128 + wm * 32;
  const int nBase = blockIdx.x * 128 + wn * 64;

  const __bf16* aRow[2];
  const __bf16* bRow[4];
#pragma unroll
  for (int mi = 0; mi < 2; ++mi) aRow[mi] = A + (size_t)(mBase + mi * 16 + col) * K;
#pragma unroll
  for (int ni = 0; ni < 4; ++ni) bRow[ni] = BT + (size_t)(nBase + ni * 16 + col) * K;

  f32x8 acc[2][4];
#pragma unroll
  for (int mi = 0; mi < 2; ++mi)
#pragma unroll
    for (int ni = 0; ni < 4; ++ni)
#pragma unroll
      for (int e = 0; e < 8; ++e) acc[mi][ni][e] = 0.0f;

  for (int k = 0; k < K; k += 32) {
    bf16x16 af[2], bfr[4];
#pragma unroll
    for (int mi = 0; mi < 2; ++mi) af[mi] = load_a_frag(aRow[mi], k, hi);
#pragma unroll
    for (int ni = 0; ni < 4; ++ni) bfr[ni] = load_b_frag(bRow[ni], k, hi);
#pragma unroll
    for (int mi = 0; mi < 2; ++mi)
#pragma unroll
      for (int ni = 0; ni < 4; ++ni)
        acc[mi][ni] = wmma_bf16(af[mi], bfr[ni], acc[mi][ni]);
  }

#pragma unroll
  for (int mi = 0; mi < 2; ++mi)
#pragma unroll
    for (int ni = 0; ni < 4; ++ni)
#pragma unroll
      for (int i = 0; i < 8; ++i) {
        int row = mBase + mi * 16 + i + 8 * hi;
        int c2 = nBase + ni * 16 + col;
        float v = acc[mi][ni][i];
        if (F32OUT)
          ((float*)outp)[(size_t)row * N + c2] = v + bias[c2];
        else
          ((__bf16*)outp)[(size_t)row * N + c2] = (__bf16)v;
      }
}

#endif  // HAS_TDM

// ---------------------------------------------------------------------------
// Windowed attention with lookback. One block per (head, window); 8 waves,
// wave wv owns query rows [wv*16, wv*16+16). Online (flash) softmax over the
// 256 KV columns in chunks of 32. Window 0's zero-padded lookback is folded
// analytically into the softmax init (m=0, l=128) and its chunks skipped.
// ---------------------------------------------------------------------------
__global__ __launch_bounds__(256) void local_attn_wmma(
    const __bf16* __restrict__ qkv,   // [N_TOK, 1536]
    __bf16* __restrict__ attnb) {     // [N_TOK, 512] token-major, head-major cols
  const int hh = blockIdx.x;   // head
  const int ww = blockIdx.y;   // window
  const int tid = threadIdx.x;
  const int lane = tid & 31;
  const int wv = tid >> 5;
  const int hi = lane >> 4;
  const int col = lane & 15;

  __shared__ __align__(16) __bf16 Vt[kDHead * 2 * kWsz];  // V^T: 64 x 256 (32 KB)
  __shared__ __align__(16) __bf16 Pst[8 * 16 * 32];       // per-wave P tile (8 KB)

  // Stage V transposed into LDS (d-major) so the PV WMMA's B fragments are
  // K-contiguous. For ww==0 the j<128 region is never read (loop starts at 128).
  {
    const int j = tid;  // 0..255 (token within the 2-window KV span)
    if (!(ww == 0 && j < 128)) {
      const int token = (ww - 1) * kWsz + j;
      const __bf16* vrow = qkv + (size_t)token * kQKV + 2 * kHD + hh * kDHead;
#pragma unroll
      for (int d0 = 0; d0 < kDHead; d0 += 8) {
        u32x4 pk = *(const u32x4*)(vrow + d0);
        const __bf16* pv = (const __bf16*)&pk;
#pragma unroll
        for (int dd = 0; dd < 8; ++dd) Vt[(d0 + dd) * (2 * kWsz) + j] = pv[dd];
      }
    }
  }
  __syncthreads();

  // Q A-fragments (16 rows x K=64 -> two K=32 fragments), loaded once.
  const int qtoken = ww * kWsz + wv * 16 + col;
  const __bf16* qrow = qkv + (size_t)qtoken * kQKV + hh * kDHead;
  const bf16x16 qa0 = load_a_frag(qrow, 0, hi);
  const bf16x16 qa1 = load_a_frag(qrow, 32, hi);

  float mrow[8], lrow[8];
  f32x8 oacc[4];
  int c0;
  if (ww == 0) {
    // 128 zero logits from the zero-padded lookback window: max>=0, sum=128.
    c0 = 4;
#pragma unroll
    for (int i = 0; i < 8; ++i) { mrow[i] = 0.0f; lrow[i] = 128.0f; }
  } else {
    c0 = 0;
#pragma unroll
    for (int i = 0; i < 8; ++i) { mrow[i] = -3.0e38f; lrow[i] = 0.0f; }
  }
#pragma unroll
  for (int s = 0; s < 4; ++s)
#pragma unroll
    for (int e = 0; e < 8; ++e) oacc[s][e] = 0.0f;

  const int kvbase = (ww - 1) * kWsz;  // token of KV column j is kvbase + j
  __bf16* pslab = &Pst[wv * 512];

  for (int c = c0; c < 8; ++c) {
    const int j0 = c * 32;

    // S chunk = Q (16x64) x K^T (64x32): 2 subtiles x 2 K-steps = 4 WMMAs.
    f32x8 sacc[2];
#pragma unroll
    for (int s = 0; s < 2; ++s)
#pragma unroll
      for (int e = 0; e < 8; ++e) sacc[s][e] = 0.0f;
#pragma unroll
    for (int s = 0; s < 2; ++s) {
      const int jcol = j0 + s * 16 + col;
      const __bf16* krow = qkv + (size_t)(kvbase + jcol) * kQKV + kHD + hh * kDHead;
      sacc[s] = wmma_bf16(qa0, load_b_frag(krow, 0, hi), sacc[s]);
      sacc[s] = wmma_bf16(qa1, load_b_frag(krow, 32, hi), sacc[s]);
    }

    // Online softmax update (rows wv*16 + i + 8*hi); half-wave reductions.
#pragma unroll
    for (int i = 0; i < 8; ++i) {
      const int rown = wv * 16 + i + 8 * hi;
      float v0 = sacc[0][i] * kScale;
      float v1 = sacc[1][i] * kScale;
      if (j0 + col > rown + kWsz) v0 = ATTN_MASK_VALUE;
      if (j0 + 16 + col > rown + kWsz) v1 = ATTN_MASK_VALUE;
      float cm = fmaxf(v0, v1);
#pragma unroll
      for (int off = 1; off < 16; off <<= 1) cm = fmaxf(cm, __shfl_xor(cm, off, 32));
      const float mnew = fmaxf(mrow[i], cm);
      const float p0 = __expf(v0 - mnew);
      const float p1 = __expf(v1 - mnew);
      float ps = p0 + p1;
#pragma unroll
      for (int off = 1; off < 16; off <<= 1) ps += __shfl_xor(ps, off, 32);
      const float sf = __expf(mrow[i] - mnew);
      lrow[i] = lrow[i] * sf + ps;
      mrow[i] = mnew;
#pragma unroll
      for (int s = 0; s < 4; ++s) oacc[s][i] *= sf;
      // Stash P (bf16) row-major 16x32 so it can be re-read in A-fragment order.
      pslab[(i + 8 * hi) * 32 + col] = (__bf16)p0;
      pslab[(i + 8 * hi) * 32 + 16 + col] = (__bf16)p1;
    }
    __syncthreads();  // LDS visibility for the C->A layout round-trip

    // O += P (16x32) x V (32x64): 4 WMMAs against LDS-transposed V.
    const bf16x16 pa = load_a_frag(&pslab[col * 32], 0, hi);
#pragma unroll
    for (int s = 0; s < 4; ++s) {
      const __bf16* vcol = &Vt[(s * 16 + col) * (2 * kWsz) + j0];
      oacc[s] = wmma_bf16(pa, load_b_frag(vcol, 0, hi), oacc[s]);
    }
    __syncthreads();  // protect Pst before next chunk's stores
  }

  // Normalize and emit bf16 in (token, h*d) layout.
#pragma unroll
  for (int i = 0; i < 8; ++i) {
    const float inv = 1.0f / lrow[i];
    const int token = ww * kWsz + wv * 16 + i + 8 * hi;
#pragma unroll
    for (int s = 0; s < 4; ++s) {
      const int ocol = hh * kDHead + s * 16 + col;
      attnb[(size_t)token * kHD + ocol] = (__bf16)(oacc[s][i] * inv);
    }
  }
}

// ---------------------------------------------------------------------------
extern "C" void kernel_launch(void* const* d_in, const int* in_sizes, int n_in,
                              void* d_out, int out_size, void* d_ws, size_t ws_size,
                              hipStream_t stream) {
  (void)in_sizes; (void)n_in; (void)out_size; (void)ws_size;

  const float* x     = (const float*)d_in[0];
  const float* w_qkv = (const float*)d_in[1];
  const float* w_out = (const float*)d_in[2];
  const float* b_out = (const float*)d_in[3];

  // Workspace layout (bf16 elements): ~105 MB total.
  __bf16* xb   = (__bf16*)d_ws;                    // 16384 x 1024
  __bf16* wqT  = xb   + (size_t)kNTok * kDim;      // 1536 x 1024 (w_qkv^T)
  __bf16* woT  = wqT  + (size_t)kQKV * kDim;       // 1024 x 512  (w_out^T)
  __bf16* qkvb = woT  + (size_t)kDim * kHD;        // 16384 x 1536
  __bf16* attb = qkvb + (size_t)kNTok * kQKV;      // 16384 x 512

  cvt_f32_to_bf16<<<4096, 256, 0, stream>>>(x, xb, kNTok * kDim);
  transpose_f32_to_bf16<<<1024, 256, 0, stream>>>(w_qkv, wqT, kDim, kQKV);
  transpose_f32_to_bf16<<<512, 256, 0, stream>>>(w_out, woT, kHD, kDim);

  dim3 gQKV(kQKV / 128, kNTok / 128);  // 12 x 128
  gemm_bf16_wmma<false><<<gQKV, 256, 0, stream>>>(xb, wqT, nullptr, qkvb, kQKV, kDim);

  dim3 gAttn(kHeads, kNWin);           // 8 x 128
  local_attn_wmma<<<gAttn, 256, 0, stream>>>(qkvb, attb);

  dim3 gOut(kDim / 128, kNTok / 128);  // 8 x 128
  gemm_bf16_wmma<true><<<gOut, 256, 0, stream>>>(attb, woT, b_out, d_out, kDim, kHD);
}
